// LoFTREncoderLayer_8211977470152
// MI455X (gfx1250) — compile-verified, matrix operand order checked by reading, not run on hardware
//
#include <hip/hip_runtime.h>
#include <hip/hip_bf16.h>
#include <math.h>

// ---------------------------------------------------------------------------
// LoFTR encoder layer (linear attention), gfx1250.
// f16 WMMA GEMMs (v_wmma_f32_16x16x32_f16), double-buffered TDM-staged LDS
// tiles (tensor_load_to_lds + s_wait_tensorcnt), b128 fragment loads,
// head-major transposed K'/V' so the KV stage has zero strided gathers.
// D=256, H=8, Dh=32, N=8, L=S=8192, R=65536.
// ---------------------------------------------------------------------------

typedef __attribute__((ext_vector_type(16))) _Float16 v16h;
typedef __attribute__((ext_vector_type(8)))  float    v8f;
typedef __attribute__((ext_vector_type(4)))  unsigned int u32x4;
typedef __attribute__((ext_vector_type(8)))  int       i32x8;
typedef __attribute__((ext_vector_type(4)))  int       i32x4;

#define DMODEL   256
#define NHEAD    8
#define HEADDIM  32
#define NBATCH   8
#define SEQLEN   8192
#define RTOT     (NBATCH * SEQLEN)   // 65536 rows

#if __has_builtin(__builtin_amdgcn_tensor_load_to_lds)
#define HAS_TDM 1
#else
#define HAS_TDM 0
#endif

union F16Frag { v16h v; uint4 q[2]; };

__device__ __forceinline__ v8f wmma_f16(v16h a, v16h b, v8f c) {
  return __builtin_amdgcn_wmma_f32_16x16x32_f16(false, a, false, b, (short)0, c,
                                                false, false);
}

#if HAS_TDM
// One TDM 2-D tile load: tile_k contiguous f16 elems x tile_rows, row stride
// row_stride_elems, into LDS at lds_byte_off (rows packed, tile_k per row).
// D# layout per CDNA5 ISA 8.3/8.4.
__device__ __forceinline__ void tdm_load_2d(unsigned lds_byte_off,
                                            const _Float16* gptr,
                                            unsigned tile_k, unsigned tile_rows,
                                            unsigned long long row_stride_elems) {
  unsigned long long ga = (unsigned long long)(size_t)gptr;
  u32x4 g0;
  g0[0] = 1u;                                    // count=1 (valid user D#)
  g0[1] = lds_byte_off;                          // lds_addr
  g0[2] = (unsigned)ga;                          // global_addr[31:0]
  g0[3] = (unsigned)((ga >> 32) & 0x01FFFFFFu)   // global_addr[56:32]
          | 0x80000000u;                         // type=2 ("image")
  const unsigned dim0 = 0x40000000u;             // huge tensor dims: no OOB clip
  const unsigned dim1 = 0x40000000u;
  i32x8 g1;
  g1[0] = (int)(1u << 16);                       // data_size=1 (2B), mask=0
  g1[1] = (int)((dim0 & 0xFFFFu) << 16);         // tensor_dim0[15:0]
  g1[2] = (int)((dim0 >> 16) | ((dim1 & 0xFFFFu) << 16));
  g1[3] = (int)((dim1 >> 16) | (tile_k << 16));  // dim1 hi | tile_dim0
  g1[4] = (int)(tile_rows);                      // tile_dim1 (tile_dim2=0)
  g1[5] = (int)(unsigned)(row_stride_elems & 0xFFFFFFFFull);
  g1[6] = (int)(unsigned)((row_stride_elems >> 32) & 0xFFFFull);
  g1[7] = 0;
  i32x4 z4 = {0, 0, 0, 0};
#if __clang_major__ >= 23
  i32x8 z8 = {0, 0, 0, 0, 0, 0, 0, 0};
  __builtin_amdgcn_tensor_load_to_lds(g0, g1, z4, z4, z8, 0);
#else
  __builtin_amdgcn_tensor_load_to_lds(g0, g1, z4, z4, 0);
#endif
}
#endif  // HAS_TDM

// activation codes
#define ACT_NONE  0
#define ACT_ELU1  1   // elu(x)+1 == x>0 ? x+1 : exp(x)
#define ACT_RELU  2
#define ACT_SCALE 3

__device__ __forceinline__ float apply_act(float v, int act, float scale) {
  if (act == ACT_ELU1)  return v > 0.f ? v + 1.f : __expf(v);
  if (act == ACT_RELU)  return fmaxf(v, 0.f);
  if (act == ACT_SCALE) return v * scale;
  return v;
}

// ---------------------------------------------------------------------------
// elementwise / layout helpers
// ---------------------------------------------------------------------------
__global__ void cvt_f32_f16_kernel(const float* __restrict__ s,
                                   _Float16* __restrict__ d, int n) {
  int i = blockIdx.x * blockDim.x + threadIdx.x;
  if (i < n) d[i] = (_Float16)s[i];
}

// W [K,256] f32  ->  Wt [256,K] f16  (K-major weight for B-fragment loads)
__global__ __launch_bounds__(256) void transpose_w_kernel(const float* __restrict__ W,
                                                          _Float16* __restrict__ Wt,
                                                          int K) {
  int k = blockIdx.x;
  int n = threadIdx.x;
  Wt[(size_t)n * K + k] = (_Float16)W[(size_t)k * DMODEL + n];
}

// f16(x) into cat[:, 0:256] (row stride 512)
__global__ __launch_bounds__(256) void pack_x_kernel(const float* __restrict__ x,
                                                     _Float16* __restrict__ cat) {
  int row = blockIdx.x;
  int t   = threadIdx.x;
  cat[(size_t)row * 512 + t] = (_Float16)x[(size_t)row * DMODEL + t];
}

// ---------------------------------------------------------------------------
// WMMA GEMM:  out[R,256] = act( A[R,Kdim] @ W[Kdim,256] )   (W given K-major)
// Block 64x64 tile, 256 threads = 8 waves (4 M x 2 N), K-chunks of 32.
// Double-buffered TDM staging: issue tile i+1, then s_wait_tensorcnt 2 so the
// next tile's two DMAs stay in flight while tile i is consumed.
// Outputs: outh (f16 row-major), outf (f32 row-major), outT (f16 head-major
// transposed [n][col][SEQLEN], via LDS transpose stage -> coalesced b128).
// ---------------------------------------------------------------------------
__global__ __launch_bounds__(256)
void gemm_kernel(const _Float16* __restrict__ A, int ldA,
                 const _Float16* __restrict__ Wt, int Kdim,
                 _Float16* __restrict__ outh, float* __restrict__ outf,
                 _Float16* __restrict__ outT, int act, float scale) {
  __shared__ __align__(64) _Float16 As[2][64 * 32];
  __shared__ __align__(64) _Float16 Bst[2][64 * 32];
  __shared__ __align__(64) _Float16 Ts[64 * 72];    // padded transpose stage

  const int mb   = blockIdx.x * 64;
  const int nb   = blockIdx.y * 64;
  const int tid  = threadIdx.x;
  const int lane = tid & 31;
  const int w    = tid >> 5;
  const int wm   = w >> 1;                 // 0..3
  const int wn   = w & 1;                  // 0..1
  const int m    = lane & 15;
  const int kb   = (lane >> 4) << 3;       // 0 or 8
  const int hi   = (lane >> 4) << 3;       // C-layout row offset

  const int nIter = Kdim >> 5;
  v8f acc0 = {};
  v8f acc1 = {};

#if HAS_TDM
  if (tid == 0) {
    tdm_load_2d((unsigned)(size_t)(void*)&As[0][0],
                A + (size_t)mb * ldA, 32u, 64u, (unsigned long long)ldA);
    tdm_load_2d((unsigned)(size_t)(void*)&Bst[0][0],
                Wt + (size_t)nb * Kdim, 32u, 64u, (unsigned long long)Kdim);
  }
#endif

  for (int it = 0; it < nIter; ++it) {
    const int cur = it & 1;
#if HAS_TDM
    if (tid == 0) {
      if (it + 1 < nIter) {
        int k1 = (it + 1) << 5;
        tdm_load_2d((unsigned)(size_t)(void*)&As[cur ^ 1][0],
                    A + (size_t)mb * ldA + k1, 32u, 64u, (unsigned long long)ldA);
        tdm_load_2d((unsigned)(size_t)(void*)&Bst[cur ^ 1][0],
                    Wt + (size_t)nb * Kdim + k1, 32u, 64u,
                    (unsigned long long)Kdim);
        __builtin_amdgcn_s_wait_tensorcnt(2);   // tile `it` complete
      } else {
        __builtin_amdgcn_s_wait_tensorcnt(0);   // drain
      }
    }
#else
    {
      int k0 = it << 5;
      int r = tid >> 2, c = (tid & 3) * 8;
      *(uint4*)&As[cur][r * 32 + c] =
          *(const uint4*)(A + (size_t)(mb + r) * ldA + k0 + c);
      *(uint4*)&Bst[cur][r * 32 + c] =
          *(const uint4*)(Wt + (size_t)(nb + r) * Kdim + k0 + c);
      if (k0 + 32 < Kdim) {
        __builtin_prefetch(A + (size_t)(mb + r) * ldA + k0 + 32, 0, 1);
        __builtin_prefetch(Wt + (size_t)(nb + r) * Kdim + k0 + 32, 0, 1);
      }
    }
#endif
    __syncthreads();   // publish buf[cur]

    F16Frag a, b0, b1;
    const int rowA = (wm * 16 + m) * 32;
    a.q[0] = *(const uint4*)&As[cur][rowA + kb];
    a.q[1] = *(const uint4*)&As[cur][rowA + kb + 16];
    const int nloc = wn * 32 + (lane & 15);
    b0.q[0] = *(const uint4*)&Bst[cur][nloc * 32 + kb];
    b0.q[1] = *(const uint4*)&Bst[cur][nloc * 32 + kb + 16];
    b1.q[0] = *(const uint4*)&Bst[cur][(nloc + 16) * 32 + kb];
    b1.q[1] = *(const uint4*)&Bst[cur][(nloc + 16) * 32 + kb + 16];

    acc0 = wmma_f16(a.v, b0.v, acc0);
    acc1 = wmma_f16(a.v, b1.v, acc1);
    __syncthreads();   // all waves done with buf[cur] before it is re-filled
  }

  if (outh || outf) {
#pragma unroll
    for (int i = 0; i < 8; ++i) {
      int r = mb + wm * 16 + i + hi;
      int c = nb + wn * 32 + (lane & 15);
      float v0 = apply_act(acc0[i], act, scale);
      float v1 = apply_act(acc1[i], act, scale);
      if (outh) {
        outh[(size_t)r * DMODEL + c]      = (_Float16)v0;
        outh[(size_t)r * DMODEL + c + 16] = (_Float16)v1;
      }
      if (outf) {
        outf[(size_t)r * DMODEL + c]      = v0;
        outf[(size_t)r * DMODEL + c + 16] = v1;
      }
    }
  }

  if (outT) {
    // stage transposed tile in LDS, then coalesced 32B stores along s
#pragma unroll
    for (int i = 0; i < 8; ++i) {
      int rloc = wm * 16 + i + hi;           // s within tile
      int c0   = wn * 32 + (lane & 15);      // col within tile
      Ts[c0 * 72 + rloc]        = (_Float16)apply_act(acc0[i], act, scale);
      Ts[(c0 + 16) * 72 + rloc] = (_Float16)apply_act(acc1[i], act, scale);
    }
    __syncthreads();
    int dloc = tid >> 2;                     // 0..63 (tile col = head-major d)
    int sloc = (tid & 3) * 16;               // 0..48 (s within tile)
    uint4 u0 = *(const uint4*)&Ts[dloc * 72 + sloc];
    uint4 u1 = *(const uint4*)&Ts[dloc * 72 + sloc + 8];
    int nidx = mb >> 13;                     // 8192 rows per batch
    size_t dst = ((size_t)nidx * DMODEL + nb + dloc) * SEQLEN + (mb & 8191) + sloc;
    *(uint4*)&outT[dst]     = u0;
    *(uint4*)&outT[dst + 8] = u1;
  }
}

// ---------------------------------------------------------------------------
// KVhT[n,h][v][d] = sum_s V'[n,s,h,v] * K'[n,s,h,d]
// K'/V' arrive head-major transposed ([n][h*32+d][S]) so all fragments are
// contiguous b128 loads.  One workgroup per (n,h); 4 waves, 16x16 quadrants.
// ---------------------------------------------------------------------------
__global__ __launch_bounds__(128)
void kv_kernel(const _Float16* __restrict__ KpT, const _Float16* __restrict__ VpT,
               _Float16* __restrict__ KVhT) {
  const int nh   = blockIdx.x;
  const int w    = threadIdx.x >> 5;
  const int lane = threadIdx.x & 31;
  const int vq   = (w >> 1) * 16;   // v quadrant (rows of C)
  const int dq   = (w & 1) * 16;    // d quadrant (cols of C)
  const int m    = lane & 15;
  const int kb   = (lane >> 4) << 3;
  const int hi   = (lane >> 4) << 3;

  const _Float16* Va = VpT + ((size_t)nh * HEADDIM + vq + m) * SEQLEN;
  const _Float16* Kb = KpT + ((size_t)nh * HEADDIM + dq + (lane & 15)) * SEQLEN;

  v8f acc = {};
  for (int s0 = 0; s0 < SEQLEN; s0 += 32) {
    F16Frag a, b;
    a.q[0] = *(const uint4*)&Va[s0 + kb];        // A[v][s] = V'[s,v]
    a.q[1] = *(const uint4*)&Va[s0 + kb + 16];
    b.q[0] = *(const uint4*)&Kb[s0 + kb];        // B[s][d] = K'[s,d]
    b.q[1] = *(const uint4*)&Kb[s0 + kb + 16];
    acc = wmma_f16(a.v, b.v, acc);
  }

#pragma unroll
  for (int i = 0; i < 8; ++i) {
    int v = vq + i + hi;
    int d = dq + (lane & 15);
    KVhT[(size_t)nh * 1024 + v * 32 + d] = (_Float16)acc[i];
  }
}

// Ksum[n,h,d] = sum_s K'[n,s,h,d]  — vectorized row sums over KpT
__global__ __launch_bounds__(256)
void ksum_kernel(const _Float16* __restrict__ KpT, float* __restrict__ Ksum) {
  __shared__ float part[32][8];
  const int nh  = blockIdx.x;
  const int d   = threadIdx.x >> 3;   // 0..31
  const int seg = threadIdx.x & 7;    // 0..7
  const _Float16* base =
      KpT + ((size_t)nh * HEADDIM + d) * SEQLEN + seg * (SEQLEN / 8);
  float s = 0.f;
  for (int i = 0; i < SEQLEN / 8; i += 8) {
    uint4 u = *(const uint4*)&base[i];
    const _Float16* hp = (const _Float16*)&u;
#pragma unroll
    for (int j = 0; j < 8; ++j) s += (float)hp[j];
  }
  part[d][seg] = s;
  __syncthreads();
  if (seg == 0) {
    float t = 0.f;
#pragma unroll
    for (int j = 0; j < 8; ++j) t += part[d][j];
    Ksum[nh * HEADDIM + d] = t;
  }
}

// ---------------------------------------------------------------------------
// msg0[n,l,h,:] = (Q[n,l,h,:] @ KV[n,h]) * Z[n,l,h]   (Z folds *S and eps)
// One block = 16 L rows; 8 waves = 8 heads; K=32 exactly -> 2 wmma per wave.
// ---------------------------------------------------------------------------
__global__ __launch_bounds__(256)
void attn_out_kernel(const _Float16* __restrict__ Qp,
                     const _Float16* __restrict__ KVhT,
                     const float* __restrict__ Ksum,
                     _Float16* __restrict__ msg0) {
  __shared__ __align__(64) _Float16 Qs[16 * 256];
  __shared__ float Zs[8][16];

  const int rb   = blockIdx.x * 16;
  const int tid  = threadIdx.x;
  const int w    = tid >> 5;          // head
  const int lane = tid & 31;
  const int n    = rb / SEQLEN;
  const int h    = w;
  const int kb   = (lane >> 4) << 3;
  const int m    = lane & 15;
  const int hi   = (lane >> 4) << 3;

  {
    int r = tid >> 4, c = (tid & 15) * 16;
    const uint4* srcp = (const uint4*)(Qp + (size_t)(rb + r) * DMODEL + c);
    *(uint4*)&Qs[r * 256 + c]     = srcp[0];
    *(uint4*)&Qs[r * 256 + c + 8] = srcp[1];
  }
  __syncthreads();

  if (lane < 16) {
    const float* ks = Ksum + (n * NHEAD + h) * HEADDIM;
    float dsum = 0.f;
#pragma unroll
    for (int d = 0; d < HEADDIM; ++d)
      dsum += (float)Qs[lane * 256 + h * HEADDIM + d] * ks[d];
    Zs[w][lane] = (float)SEQLEN / (dsum + 1e-6f);
  }
  __syncthreads();

  F16Frag a, b0, b1;
  a.q[0] = *(const uint4*)&Qs[m * 256 + h * HEADDIM + kb];
  a.q[1] = *(const uint4*)&Qs[m * 256 + h * HEADDIM + kb + 16];
  const _Float16* kvt = KVhT + (size_t)(n * NHEAD + h) * 1024;
  b0.q[0] = *(const uint4*)&kvt[(lane & 15) * 32 + kb];       // B[d][v]=KVhT[v][d]
  b0.q[1] = *(const uint4*)&kvt[(lane & 15) * 32 + kb + 16];
  b1.q[0] = *(const uint4*)&kvt[((lane & 15) + 16) * 32 + kb];
  b1.q[1] = *(const uint4*)&kvt[((lane & 15) + 16) * 32 + kb + 16];

  v8f c0 = {};
  v8f c1 = {};
  c0 = wmma_f16(a.v, b0.v, c0);
  c1 = wmma_f16(a.v, b1.v, c1);

#pragma unroll
  for (int i = 0; i < 8; ++i) {
    int r = i + hi;
    float z = Zs[w][r];
    msg0[(size_t)(rb + r) * DMODEL + h * HEADDIM + (lane & 15)]      = (_Float16)(c0[i] * z);
    msg0[(size_t)(rb + r) * DMODEL + h * HEADDIM + 16 + (lane & 15)] = (_Float16)(c1[i] * z);
  }
}

// ---------------------------------------------------------------------------
// LayerNorm kernels (one 256-col row per 256-thread block, LDS tree reduce)
// ---------------------------------------------------------------------------
__global__ __launch_bounds__(256)
void ln_to_cat_kernel(const float* __restrict__ tin, const float* __restrict__ g,
                      const float* __restrict__ b, _Float16* __restrict__ cat) {
  __shared__ float red[256];
  const int row = blockIdx.x;
  const int t   = threadIdx.x;
  float v = tin[(size_t)row * DMODEL + t];

  red[t] = v; __syncthreads();
  for (int off = 128; off > 0; off >>= 1) {
    if (t < off) red[t] += red[t + off];
    __syncthreads();
  }
  float mu = red[0] * (1.f / DMODEL);
  __syncthreads();

  float dv = v - mu;
  red[t] = dv * dv; __syncthreads();
  for (int off = 128; off > 0; off >>= 1) {
    if (t < off) red[t] += red[t + off];
    __syncthreads();
  }
  float var = red[0] * (1.f / DMODEL);

  float y = dv * rsqrtf(var + 1e-5f) * g[t] + b[t];
  cat[(size_t)row * 512 + DMODEL + t] = (_Float16)y;
}

__global__ __launch_bounds__(256)
void ln_residual_kernel(const float* __restrict__ tin, const float* __restrict__ x,
                        const float* __restrict__ g, const float* __restrict__ b,
                        float* __restrict__ out) {
  __shared__ float red[256];
  const int row = blockIdx.x;
  const int t   = threadIdx.x;
  float v = tin[(size_t)row * DMODEL + t];

  red[t] = v; __syncthreads();
  for (int off = 128; off > 0; off >>= 1) {
    if (t < off) red[t] += red[t + off];
    __syncthreads();
  }
  float mu = red[0] * (1.f / DMODEL);
  __syncthreads();

  float dv = v - mu;
  red[t] = dv * dv; __syncthreads();
  for (int off = 128; off > 0; off >>= 1) {
    if (t < off) red[t] += red[t + off];
    __syncthreads();
  }
  float var = red[0] * (1.f / DMODEL);

  float y = dv * rsqrtf(var + 1e-5f) * g[t] + b[t];
  out[(size_t)row * DMODEL + t] = x[(size_t)row * DMODEL + t] + y;
}

// ---------------------------------------------------------------------------
// host-side orchestration
// ---------------------------------------------------------------------------
extern "C" void kernel_launch(void* const* d_in, const int* in_sizes, int n_in,
                              void* d_out, int out_size, void* d_ws, size_t ws_size,
                              hipStream_t stream) {
  const float* x   = (const float*)d_in[0];
  const float* src = (const float*)d_in[1];
  const float* Wq  = (const float*)d_in[2];
  const float* Wk  = (const float*)d_in[3];
  const float* Wv  = (const float*)d_in[4];
  const float* Wm  = (const float*)d_in[5];
  const float* W1  = (const float*)d_in[6];
  const float* W2  = (const float*)d_in[7];
  const float* g1  = (const float*)d_in[8];
  const float* b1  = (const float*)d_in[9];
  const float* g2  = (const float*)d_in[10];
  const float* b2  = (const float*)d_in[11];
  float* out = (float*)d_out;

  char* base = (char*)d_ws;
  size_t off = 0;
  auto alloc = [&](size_t bytes) -> char* {
    char* p = base + off;
    off = (off + bytes + 255) & ~(size_t)255;
    return p;
  };
  const size_t R = RTOT;
  _Float16* hWtq = (_Float16*)alloc((size_t)DMODEL * DMODEL * 2);     // K-major
  _Float16* hWtk = (_Float16*)alloc((size_t)DMODEL * DMODEL * 2);
  _Float16* hWtv = (_Float16*)alloc((size_t)DMODEL * DMODEL * 2);
  _Float16* hWtm = (_Float16*)alloc((size_t)DMODEL * DMODEL * 2);
  _Float16* hWt1 = (_Float16*)alloc((size_t)DMODEL * 2 * DMODEL * 2); // [256][512]
  _Float16* hWt2 = (_Float16*)alloc((size_t)DMODEL * DMODEL * 2);
  _Float16* srch = (_Float16*)alloc(R * DMODEL * 2);
  _Float16* Qp   = (_Float16*)alloc(R * DMODEL * 2);   // reused as FFN hidden
  _Float16* KpT  = (_Float16*)alloc(R * DMODEL * 2);   // [n][h*32+d][S]
  _Float16* VpT  = (_Float16*)alloc(R * DMODEL * 2);   // [n][h*32+v][S]
  _Float16* msg0 = (_Float16*)alloc(R * DMODEL * 2);
  _Float16* cat  = (_Float16*)alloc(R * 2 * DMODEL * 2);
  float*    tmp  = (float*)   alloc(R * DMODEL * 4);
  _Float16* KVhT = (_Float16*)alloc((size_t)NBATCH * NHEAD * 1024 * 2);
  float*    Ksum = (float*)   alloc((size_t)NBATCH * NHEAD * HEADDIM * 4);
  (void)ws_size; (void)n_in; (void)in_sizes; (void)out_size;

  // 1) weights: f32 -> f16, transposed to K-major
  transpose_w_kernel<<<DMODEL, 256, 0, stream>>>(Wq, hWtq, DMODEL);
  transpose_w_kernel<<<DMODEL, 256, 0, stream>>>(Wk, hWtk, DMODEL);
  transpose_w_kernel<<<DMODEL, 256, 0, stream>>>(Wv, hWtv, DMODEL);
  transpose_w_kernel<<<DMODEL, 256, 0, stream>>>(Wm, hWtm, DMODEL);
  transpose_w_kernel<<<2 * DMODEL, 256, 0, stream>>>(W1, hWt1, 2 * DMODEL);
  transpose_w_kernel<<<DMODEL, 256, 0, stream>>>(W2, hWt2, DMODEL);

  // 2) activations to f16: x into cat[:,0:256]; source into srch
  pack_x_kernel<<<R, 256, 0, stream>>>(x, cat);
  cvt_f32_f16_kernel<<<(int)(R * DMODEL / 256), 256, 0, stream>>>(src, srch,
                                                                  (int)(R * DMODEL));

  // 3) projections: Q' row-major; K'/V' head-major transposed
  dim3 ggrid(R / 64, DMODEL / 64);
  gemm_kernel<<<ggrid, 256, 0, stream>>>(cat, 512, hWtq, DMODEL, Qp,
                                         (float*)nullptr, (_Float16*)nullptr,
                                         ACT_ELU1, 1.f);
  gemm_kernel<<<ggrid, 256, 0, stream>>>(srch, 256, hWtk, DMODEL,
                                         (_Float16*)nullptr, (float*)nullptr,
                                         KpT, ACT_ELU1, 1.f);
  gemm_kernel<<<ggrid, 256, 0, stream>>>(srch, 256, hWtv, DMODEL,
                                         (_Float16*)nullptr, (float*)nullptr,
                                         VpT, ACT_SCALE, 1.f / (float)SEQLEN);

  // 4) KV^T (f16) and Ksum
  kv_kernel<<<NBATCH * NHEAD, 128, 0, stream>>>(KpT, VpT, KVhT);
  ksum_kernel<<<NBATCH * NHEAD, 256, 0, stream>>>(KpT, Ksum);

  // 5) attention output
  attn_out_kernel<<<R / 16, 256, 0, stream>>>(Qp, KVhT, Ksum, msg0);

  // 6) msg0 @ Wm -> tmp (f32)
  gemm_kernel<<<ggrid, 256, 0, stream>>>(msg0, 256, hWtm, DMODEL,
                                         (_Float16*)nullptr, tmp,
                                         (_Float16*)nullptr, ACT_NONE, 1.f);

  // 7) LayerNorm(tmp; g1,b1) -> cat[:,256:512]
  ln_to_cat_kernel<<<R, 256, 0, stream>>>(tmp, g1, b1, cat);

  // 8) relu(cat @ W1) -> hidden (reuse Qp)
  gemm_kernel<<<ggrid, 256, 0, stream>>>(cat, 512, hWt1, 2 * DMODEL, Qp,
                                         (float*)nullptr, (_Float16*)nullptr,
                                         ACT_RELU, 1.f);

  // 9) hidden @ W2 -> tmp (f32)
  gemm_kernel<<<ggrid, 256, 0, stream>>>(Qp, 256, hWt2, DMODEL,
                                         (_Float16*)nullptr, tmp,
                                         (_Float16*)nullptr, ACT_NONE, 1.f);

  // 10) out = x + LayerNorm(tmp; g2,b2)
  ln_residual_kernel<<<R, 256, 0, stream>>>(tmp, x, g2, b2, out);
}
